// DscaLSTMCell_19172734009719
// MI455X (gfx1250) — compile-verified, matrix operand order checked by reading, not run on hardware
//
#include <hip/hip_runtime.h>
#include <hip/hip_bf16.h>

typedef __attribute__((ext_vector_type(16))) __bf16 v16bf;
typedef __attribute__((ext_vector_type(8)))  float  v8f;
typedef unsigned short u16;
typedef unsigned int   u32;
typedef unsigned long long u64;

// Problem dims (fixed by reference)
#define BB    2048   // batch
#define UU    1024
#define MM    2048   // E1+E2
#define FOURU 4096

__device__ __forceinline__ u16 f32_to_bf16_rne(float f) {
  u32 u = __float_as_uint(f);
  u32 r = u + 0x7FFFu + ((u >> 16) & 1u);
  return (u16)(r >> 16);
}

// ---------------------------------------------------------------------------
// f32 -> bf16 conversion (grid-stride), for A-side (activation) operands.
// ---------------------------------------------------------------------------
__global__ void convert_f32_to_bf16(const float* __restrict__ src,
                                    u16* __restrict__ dst, long long n) {
  long long i = (long long)blockIdx.x * blockDim.x + threadIdx.x;
  long long stride = (long long)gridDim.x * blockDim.x;
  for (; i < n; i += stride) dst[i] = f32_to_bf16_rne(src[i]);
}

// ---------------------------------------------------------------------------
// f32 -> bf16 convert + transpose for weights (B-side operands).
// src: K x N row-major f32;  dst: N x K row-major bf16 ("Bt").
// LDS-tiled 32x32 so both global accesses stay coalesced. One-time cost.
// ---------------------------------------------------------------------------
__global__ __launch_bounds__(256)
void convert_transpose_f32_bf16(const float* __restrict__ src,
                                u16* __restrict__ dst, int K, int N) {
  __shared__ float t[32][33];
  const int n0 = blockIdx.x * 32;
  const int k0 = blockIdx.y * 32;
  const int tx = threadIdx.x & 31;
  const int ty = threadIdx.x >> 5;   // 0..7
#pragma unroll
  for (int j = 0; j < 4; ++j) {
    int k = ty + j * 8;
    t[k][tx] = src[(size_t)(k0 + k) * N + (n0 + tx)];
  }
  __syncthreads();
#pragma unroll
  for (int j = 0; j < 4; ++j) {
    int n = ty + j * 8;
    dst[(size_t)(n0 + n) * K + (k0 + tx)] = f32_to_bf16_rne(t[tx][n]);
  }
}

// ---------------------------------------------------------------------------
// Tiled bf16 GEMM, f32 accumulate, v_wmma_f32_16x16x32_bf16.
// Block tile 128(M) x 256(N), BK=32, 256 threads = 8 waves (2Mx4N),
// each wave computes 64x64 = 4x4 fragments -> 16 WMMAs per K-tile.
// A : M x K row-major bf16 (lda)
// Bt: N x K row-major bf16 (ldbt == K)  [weights pre-transposed]
// C : M x N f32 (ldc)
// Staging: double-buffered LDS fed by global_load_async_to_lds_b128
// (ASYNCcnt), so tile t+1 streams in while tile t is in the matrix cores.
// ---------------------------------------------------------------------------
union FragBF { v16bf v; int4 q[2]; };

__global__ __launch_bounds__(256)
void gemm_bf16_wmma(const u16* __restrict__ A, const u16* __restrict__ Bt,
                    float* __restrict__ C, int M, int N, int K,
                    int lda, int ldbt, int ldc) {
  __shared__ __align__(16) u16 sA[2][128 * 32];  // [m][k]
  __shared__ __align__(16) u16 sB[2][256 * 32];  // [n][k]

  const int tid     = threadIdx.x;
  const int lane    = tid & 31;
  const int laneLow = lane & 15;
  const int laneHi  = lane >> 4;
  const int waveId  = tid >> 5;
  const int waveM   = waveId & 1;    // 2 waves along M (64 rows each)
  const int waveN   = waveId >> 1;   // 4 waves along N (64 cols each)
  const int mBase   = blockIdx.y * 128;
  const int nBase   = blockIdx.x * 256;
  const int mOff    = waveM * 64;
  const int nOff    = waveN * 64;

  v8f acc[4][4] = {};

  // Async-stage one K-tile (A: 128x32, B: 256x32) into LDS buffer `buf`.
  auto stage = [&](int buf, int k0) {
    // A tile: 512 x 16B chunks, 2 per thread
#pragma unroll
    for (int it = 0; it < 2; ++it) {
      int idx = tid + it * 256;
      int r   = idx >> 2;             // 0..127
      int c   = (idx & 3) << 3;       // 0,8,16,24
      u64 g   = (u64)(const void*)(A + (size_t)(mBase + r) * lda + (k0 + c));
      u32 l   = (u32)(size_t)(&sA[buf][r * 32 + c]);
      asm volatile("global_load_async_to_lds_b128 %0, %1, off"
                   :: "v"(l), "v"(g) : "memory");
    }
    // B tile: 1024 x 16B chunks, 4 per thread
#pragma unroll
    for (int it = 0; it < 4; ++it) {
      int idx = tid + it * 256;
      int r   = idx >> 2;             // 0..255
      int c   = (idx & 3) << 3;
      u64 g   = (u64)(const void*)(Bt + (size_t)(nBase + r) * ldbt + (k0 + c));
      u32 l   = (u32)(size_t)(&sB[buf][r * 32 + c]);
      asm volatile("global_load_async_to_lds_b128 %0, %1, off"
                   :: "v"(l), "v"(g) : "memory");
    }
  };

  stage(0, 0);
  asm volatile("s_wait_asynccnt 0x0" ::: "memory");
  __syncthreads();

  const int nT = K >> 5;
  for (int kt = 0; kt < nT; ++kt) {
    const int buf = kt & 1;
    if (kt + 1 < nT) stage(buf ^ 1, (kt + 1) << 5);

    // Fragments per documented 16-bit matrix VGPR layout:
    // lanes 0-15: K 0-7 then 16-23 ; lanes 16-31: K 8-15 then 24-31.
    FragBF af[4], bfr[4];
#pragma unroll
    for (int f = 0; f < 4; ++f) {
      int row  = mOff + f * 16 + laneLow;
      int base = row * 32 + laneHi * 8;
      af[f].q[0] = *(const int4*)(&sA[buf][base]);
      af[f].q[1] = *(const int4*)(&sA[buf][base + 16]);
    }
#pragma unroll
    for (int g = 0; g < 4; ++g) {
      int col  = nOff + g * 16 + laneLow;
      int base = col * 32 + laneHi * 8;
      bfr[g].q[0] = *(const int4*)(&sB[buf][base]);
      bfr[g].q[1] = *(const int4*)(&sB[buf][base + 16]);
    }

#pragma unroll
    for (int f = 0; f < 4; ++f)
#pragma unroll
      for (int g = 0; g < 4; ++g)
        acc[f][g] = __builtin_amdgcn_wmma_f32_16x16x32_bf16(
            false, af[f].v, false, bfr[g].v, (short)0, acc[f][g], false, false);

    // next tile's async copies must land before the buffer swap
    asm volatile("s_wait_asynccnt 0x0" ::: "memory");
    __syncthreads();
  }

  // Epilogue: C layout: VGPR r -> M=r (+8 for lanes 16-31), N=laneLow.
#pragma unroll
  for (int f = 0; f < 4; ++f) {
#pragma unroll
    for (int g = 0; g < 4; ++g) {
      int col = nBase + nOff + g * 16 + laneLow;
#pragma unroll
      for (int r = 0; r < 8; ++r) {
        int row = mBase + mOff + f * 16 + r + 8 * laneHi;
        C[(size_t)row * ldc + col] = acc[f][g][r];
      }
    }
  }
}

// ---------------------------------------------------------------------------
// Fused attention: s = tanh(t1+t2)*t3 ; p = softmax_row(s) ; out = e * p
// One block per row (1024 cols, 256 threads x 4). Writes f32 x and bf16 x.
// ---------------------------------------------------------------------------
__global__ __launch_bounds__(256)
void attention_kernel(const float* __restrict__ t1, const float* __restrict__ t2,
                      const float* __restrict__ t3, const float* __restrict__ e,
                      float* __restrict__ x, u16* __restrict__ xbf, int colOff) {
  const int row = blockIdx.x;
  const int tid = threadIdx.x;
  __shared__ float red[256];
  __shared__ float sv[1024];
  const size_t rb = (size_t)row * 1024;

  float s[4];
  float lmax = -3.4e38f;
#pragma unroll
  for (int j = 0; j < 4; ++j) {
    int c = tid + j * 256;
    float v = tanhf(t1[rb + c] + t2[rb + c]) * t3[rb + c];
    s[j] = v;
    lmax = fmaxf(lmax, v);
  }
  red[tid] = lmax;
  __syncthreads();
  for (int w = 128; w > 0; w >>= 1) {
    if (tid < w) red[tid] = fmaxf(red[tid], red[tid + w]);
    __syncthreads();
  }
  float rmax = red[0];
  __syncthreads();

  float lsum = 0.0f;
#pragma unroll
  for (int j = 0; j < 4; ++j) {
    int c = tid + j * 256;
    float ex = __expf(s[j] - rmax);
    sv[c] = ex;
    lsum += ex;
  }
  red[tid] = lsum;
  __syncthreads();
  for (int w = 128; w > 0; w >>= 1) {
    if (tid < w) red[tid] += red[tid + w];
    __syncthreads();
  }
  float inv = 1.0f / red[0];

  const size_t xrb = (size_t)row * MM + colOff;
#pragma unroll
  for (int j = 0; j < 4; ++j) {
    int c = tid + j * 256;
    float out = e[rb + c] * sv[c] * inv;
    x[xrb + c]   = out;
    xbf[xrb + c] = f32_to_bf16_rne(out);
  }
}

// ---------------------------------------------------------------------------
// crohis = TAU*crohis_tm*sigmoid(dcp + decomp_b) + (1-TAU)*x   (TAU = 0.5)
// ---------------------------------------------------------------------------
__global__ void crohis_kernel(const float* __restrict__ dcp,
                              const float* __restrict__ decomp_b,
                              const float* __restrict__ ctm,
                              const float* __restrict__ x,
                              float* __restrict__ out_cro,
                              u16* __restrict__ crobf, long long n) {
  long long i = (long long)blockIdx.x * blockDim.x + threadIdx.x;
  long long stride = (long long)gridDim.x * blockDim.x;
  for (; i < n; i += stride) {
    int j = (int)(i & (MM - 1));
    float d   = 1.0f / (1.0f + __expf(-(dcp[i] + decomp_b[j])));
    float cro = 0.5f * ctm[i] * d + 0.5f * x[i];
    out_cro[i] = cro;
    crobf[i]   = f32_to_bf16_rne(cro);
  }
}

// ---------------------------------------------------------------------------
// Final LSTM gates: z = z1 + z2 + bias; i,f,o=sigmoid; c=f*c_tm1+i*tanh(zc);
// h = o*tanh(c) + tanh(hc + crohis_b)
// ---------------------------------------------------------------------------
__global__ void final_gate_kernel(const float* __restrict__ z1,
                                  const float* __restrict__ z2,
                                  const float* __restrict__ bias,
                                  const float* __restrict__ c_tm1,
                                  const float* __restrict__ hc,
                                  const float* __restrict__ crohis_b,
                                  float* __restrict__ h_out,
                                  float* __restrict__ c_out, long long n) {
  long long i = (long long)blockIdx.x * blockDim.x + threadIdx.x;
  long long stride = (long long)gridDim.x * blockDim.x;
  for (; i < n; i += stride) {
    long long b = i >> 10;
    int u = (int)(i & (UU - 1));
    size_t zb = (size_t)b * FOURU;
    float zi = z1[zb + u]        + z2[zb + u]        + bias[u];
    float zf = z1[zb + UU + u]   + z2[zb + UU + u]   + bias[UU + u];
    float zc = z1[zb + 2*UU + u] + z2[zb + 2*UU + u] + bias[2*UU + u];
    float zo = z1[zb + 3*UU + u] + z2[zb + 3*UU + u] + bias[3*UU + u];
    float ig = 1.0f / (1.0f + __expf(-zi));
    float fg = 1.0f / (1.0f + __expf(-zf));
    float c  = fg * c_tm1[i] + ig * tanhf(zc);
    float og = 1.0f / (1.0f + __expf(-zo));
    float hcro = tanhf(hc[i] + crohis_b[u]);
    h_out[i] = og * tanhf(c) + hcro;
    c_out[i] = c;
  }
}

// ---------------------------------------------------------------------------
// Launcher
// ---------------------------------------------------------------------------
extern "C" void kernel_launch(void* const* d_in, const int* in_sizes, int n_in,
                              void* d_out, int out_size, void* d_ws, size_t ws_size,
                              hipStream_t stream) {
  (void)in_sizes; (void)n_in; (void)out_size; (void)ws_size;

  const float* inputs     = (const float*)d_in[0];
  const float* h_tm1      = (const float*)d_in[1];
  const float* c_tm1      = (const float*)d_in[2];
  const float* crohis_tm  = (const float*)d_in[3];
  const float* embed_w1   = (const float*)d_in[4];
  const float* embed_w2   = (const float*)d_in[5];
  const float* w12_w      = (const float*)d_in[6];
  const float* w12_u      = (const float*)d_in[7];
  const float* w12_v      = (const float*)d_in[8];
  const float* w21_w      = (const float*)d_in[9];
  const float* w21_u      = (const float*)d_in[10];
  const float* w21_v      = (const float*)d_in[11];
  const float* crohis_w   = (const float*)d_in[12];
  const float* crohis_b   = (const float*)d_in[13];
  const float* decomp_w   = (const float*)d_in[14];
  const float* decomp_b   = (const float*)d_in[15];
  const float* kernel_w   = (const float*)d_in[16];
  const float* recur_w    = (const float*)d_in[17];
  const float* bias       = (const float*)d_in[18];

  const size_t EL_BU  = (size_t)BB * UU;     // 2M
  const size_t EL_BM  = (size_t)BB * MM;     // 4M
  const size_t EL_B4U = (size_t)BB * FOURU;  // 8M

  // ---- carve workspace ----
  char* w = (char*)d_ws;
  size_t o = 0;
  auto take = [&](size_t bytes) -> void* {
    void* p = w + o;
    o += (bytes + 255) & ~(size_t)255;
    return p;
  };
  u16* XBF    = (u16*)take(EL_B4U * 2);           // inputs bf16 (B x 4096)
  u16* HBF    = (u16*)take(EL_BU * 2);            // h_tm1 bf16
  u16* CTBF   = (u16*)take(EL_BM * 2);            // crohis_tm bf16
  // pre-transposed bf16 weights: [N][K]
  u16* Wemb1T = (u16*)take((size_t)1024*2048*2);
  u16* Wemb2T = (u16*)take((size_t)1024*2048*2);
  u16* W12wT  = (u16*)take((size_t)1024*1024*2);
  u16* W12uT  = (u16*)take((size_t)1024*1024*2);
  u16* W12vT  = (u16*)take((size_t)1024*1024*2);
  u16* W21wT  = (u16*)take((size_t)1024*1024*2);
  u16* W21uT  = (u16*)take((size_t)1024*1024*2);
  u16* W21vT  = (u16*)take((size_t)1024*1024*2);
  u16* WcroT  = (u16*)take((size_t)1024*2048*2);
  u16* WdecT  = (u16*)take((size_t)2048*2048*2);
  u16* WkerT  = (u16*)take((size_t)4096*2048*2);
  u16* WrecT  = (u16*)take((size_t)4096*1024*2);
  u16* E1BF   = (u16*)take(EL_BU * 2);
  u16* E2BF   = (u16*)take(EL_BU * 2);
  u16* XOBF   = (u16*)take(EL_BM * 2);            // x bf16
  u16* CROBF  = (u16*)take(EL_BM * 2);            // crohis bf16
  float* E1F  = (float*)take(EL_BU * 4);
  float* E2F  = (float*)take(EL_BU * 4);
  float* T1   = (float*)take(EL_BU * 4);
  float* T2   = (float*)take(EL_BU * 4);
  float* T3   = (float*)take(EL_BU * 4);
  float* DCP  = (float*)take(EL_BM * 4);
  float* XF   = (float*)take(EL_BM * 4);
  float* HC   = (float*)take(EL_BU * 4);
  float* Z1   = (float*)take(EL_B4U * 4);
  float* Z2   = (float*)take(EL_B4U * 4);

  auto conv = [&](const float* s, u16* d, size_t n) {
    int blocks = (int)((n + 1023) / 1024);
    convert_f32_to_bf16<<<blocks, 256, 0, stream>>>(s, d, (long long)n);
  };
  auto convT = [&](const float* s, u16* d, int K, int N) {
    dim3 g(N / 32, K / 32);
    convert_transpose_f32_bf16<<<g, 256, 0, stream>>>(s, d, K, N);
  };
  auto gemm = [&](const u16* Ap, int lda, const u16* BtP,
                  float* Cp, int ldc, int M, int N, int K) {
    dim3 g(N / 256, M / 128);
    gemm_bf16_wmma<<<g, 256, 0, stream>>>(Ap, BtP, Cp, M, N, K, lda, K, ldc);
  };

  // ---- convert activations (A-side) + transpose-convert weights (B-side) ----
  conv(inputs,    XBF,  EL_B4U);
  conv(h_tm1,     HBF,  EL_BU);
  conv(crohis_tm, CTBF, EL_BM);
  convT(embed_w1, Wemb1T, 2048, 1024);
  convT(embed_w2, Wemb2T, 2048, 1024);
  convT(w12_w,    W12wT,  1024, 1024);
  convT(w12_u,    W12uT,  1024, 1024);
  convT(w12_v,    W12vT,  1024, 1024);
  convT(w21_w,    W21wT,  1024, 1024);
  convT(w21_u,    W21uT,  1024, 1024);
  convT(w21_v,    W21vT,  1024, 1024);
  convT(crohis_w, WcroT,  2048, 1024);
  convT(decomp_w, WdecT,  2048, 2048);
  convT(kernel_w, WkerT,  2048, 4096);
  convT(recur_w,  WrecT,  1024, 4096);

  // ---- embeds: e1 = seqx1 @ embed_w1, e2 = seqx2 @ embed_w2 ----
  gemm(XBF,        4096, Wemb1T, E1F, 1024, BB, 1024, 2048);
  gemm(XBF + 2048, 4096, Wemb2T, E2F, 1024, BB, 1024, 2048);
  conv(E1F, E1BF, EL_BU);
  conv(E2F, E2BF, EL_BU);

  // ---- attention 1->2: softmax(tanh(e1@W + e2@U) * (e2@V)) * e1 ----
  gemm(E1BF, 1024, W12wT, T1, 1024, BB, 1024, 1024);
  gemm(E2BF, 1024, W12uT, T2, 1024, BB, 1024, 1024);
  gemm(E2BF, 1024, W12vT, T3, 1024, BB, 1024, 1024);
  attention_kernel<<<BB, 256, 0, stream>>>(T1, T2, T3, E1F, XF, XOBF, 0);

  // ---- attention 2->1 ----
  gemm(E2BF, 1024, W21wT, T1, 1024, BB, 1024, 1024);
  gemm(E1BF, 1024, W21uT, T2, 1024, BB, 1024, 1024);
  gemm(E1BF, 1024, W21vT, T3, 1024, BB, 1024, 1024);
  attention_kernel<<<BB, 256, 0, stream>>>(T1, T2, T3, E2F, XF, XOBF, 1024);

  // ---- decomp + crohis update (crohis written straight into d_out) ----
  gemm(CTBF, 2048, WdecT, DCP, 2048, BB, 2048, 2048);
  float* OUT_H   = (float*)d_out;
  float* OUT_C   = OUT_H + EL_BU;
  float* OUT_CRO = OUT_H + 2 * EL_BU;
  {
    int blocks = (int)((EL_BM + 255) / 256);
    crohis_kernel<<<blocks, 256, 0, stream>>>(DCP, decomp_b, crohis_tm, XF,
                                              OUT_CRO, CROBF, (long long)EL_BM);
  }

  // ---- h_cro pre-activation, z1, z2 ----
  gemm(CROBF, 2048, WcroT, HC, 1024, BB, 1024, 2048);
  gemm(XOBF,  2048, WkerT, Z1, 4096, BB, 4096, 2048);
  gemm(HBF,   1024, WrecT, Z2, 4096, BB, 4096, 1024);

  // ---- gates + outputs ----
  {
    int blocks = (int)((EL_BU + 255) / 256);
    final_gate_kernel<<<blocks, 256, 0, stream>>>(Z1, Z2, bias, c_tm1, HC,
                                                  crohis_b, OUT_H, OUT_C,
                                                  (long long)EL_BU);
  }
}
